// AttnTextClassifier_2130303779207
// MI455X (gfx1250) — compile-verified, matrix operand order checked by reading, not run on hardware
//
#include <hip/hip_runtime.h>
#include <hip/hip_bf16.h>

// Problem constants (from reference): B=256, T=512, V=50000, E=640, D1=D2=512, C=2
#define BB   256
#define TT_  512
#define VV   50000
#define EE   640
#define D1_  512
#define D2_  512
#define K1_  (TT_ * EE)   // 327680

typedef __attribute__((ext_vector_type(16))) __bf16 v16bf;
typedef __attribute__((ext_vector_type(8)))  __bf16 v8bf;
typedef __attribute__((ext_vector_type(4)))  __bf16 v4bf;
typedef __attribute__((ext_vector_type(8)))  float  v8f;

// ---------------------------------------------------------------------------
// Kernel A: pre-convert embedding table f32 -> bf16 (once per launch,
// +192MB HBM traffic ~= 8us; removes all VALU convert work from the hot loop).
// Last block writes a zero row at index V so `* mask` becomes a row select.
// ---------------------------------------------------------------------------
__global__ void cvt_emb(const float* __restrict__ emb, __bf16* __restrict__ embbf) {
  const int nb = (VV * EE) / (256 * 4);              // 31250 full blocks
  if ((int)blockIdx.x >= nb) {
    int t = threadIdx.x;                             // zero row V: 640 elems
    if (t < EE / 4) {
      v4bf z = {};
      *(v4bf*)(embbf + (size_t)VV * EE + t * 4) = z;
    }
    return;
  }
  size_t i = ((size_t)blockIdx.x * 256 + threadIdx.x) * 4;
  const float4 v = *(const float4*)(emb + i);
  v4bf o;
  o[0] = (__bf16)v.x; o[1] = (__bf16)v.y; o[2] = (__bf16)v.z; o[3] = (__bf16)v.w;
  *(v4bf*)(embbf + i) = o;
}

// ---------------------------------------------------------------------------
// Kernel 0: seed layer-1 accumulator with bias b1 (atomics need an init value;
// d_ws is poisoned, so this must run every launch).
// ---------------------------------------------------------------------------
__global__ void init_h1(float* __restrict__ H1acc, const float* __restrict__ b1) {
  int i = blockIdx.x * blockDim.x + threadIdx.x;     // 0 .. 256*512-1
  H1acc[i] = b1[i & (D1_ - 1)];
}

// ---------------------------------------------------------------------------
// Vectorized W1 tile staging: 16 rows x 640 cols f32 -> bf16 in LDS.
// Per thread: 5x global_load_b128 (float4) + 10x cvt_pk + 5x ds_store_b64.
// ---------------------------------------------------------------------------
__device__ __forceinline__ void stage_w1(const float* __restrict__ wrow,
                                         __bf16* __restrict__ srow, int lane) {
  #pragma unroll
  for (int j = 0; j < 5; ++j) {
    const int e = (lane << 2) + (j << 7);            // lane*4 + j*128
    const float4 v = *(const float4*)(wrow + e);
    v4bf o;
    o[0] = (__bf16)v.x; o[1] = (__bf16)v.y; o[2] = (__bf16)v.z; o[3] = (__bf16)v.w;
    *(v4bf*)(srow + e) = o;
  }
}

// ---------------------------------------------------------------------------
// Kernel 1 (fast): fused embedding-gather + GEMM1, bf16 WMMA, split-K atomics.
// grid = (32 N-strips, 64 K-chunks of 8 tokens), block = 512 threads = 16 waves.
// A fragments load bf16 directly from the pre-converted table (L2-resident);
// B (W1) streams f32 from HBM exactly once, converted to bf16 via LDS staging.
// Hot loop per WMMA: 2x global b128 + 2x ds b128 + 1 v_wmma.
// ---------------------------------------------------------------------------
__global__ __launch_bounds__(512) void gemm1_fast(
    const int*    __restrict__ data,  const int*   __restrict__ mask,
    const __bf16* __restrict__ embbf, const float* __restrict__ W1,
    float*        __restrict__ H1acc) {
  __shared__ __bf16 Bs[16 * EE];            // 20 KB: [n][e], e contiguous

  const int tid  = threadIdx.x;
  const int lane = tid & 31;
  const int wave = tid >> 5;                // 0..15 -> M tile
  const int n0   = blockIdx.x << 4;         // N strip base
  const int t0   = blockIdx.y << 3;         // first token of K-chunk
  const int lm   = lane & 15;
  const int grp  = lane >> 4;               // lane group 0/1
  const int eoff = grp << 3;                // per-group K offset (A/B layout)
  const int mrow = (wave << 4) + lm;        // A-fragment row

  v8f acc = {};

  for (int t = t0; t < t0 + 8; ++t) {
    // ---- stage W1[(n0..n0+15), t*640 .. +640) as bf16 into LDS ----
    stage_w1(W1 + (size_t)(n0 + wave) * K1_ + (size_t)t * EE,
             Bs + wave * EE, lane);
    __syncthreads();

    // ---- per-lane token: mask handled by selecting the zero row ----
    const int tok = data[mrow * TT_ + t];
    const int mk  = mask[mrow * TT_ + t];
    const __bf16* erow = embbf + (size_t)(mk ? tok : VV) * EE;
    const __bf16* brow = Bs + lm * EE;

    #pragma unroll 4
    for (int es = 0; es < 20; ++es) {
      const int e0 = (es << 5) + eoff;
      const v8bf alo = *(const v8bf*)(erow + e0);
      const v8bf ahi = *(const v8bf*)(erow + e0 + 16);
      v16bf af;
      #pragma unroll
      for (int i = 0; i < 8; ++i) { af[i] = alo[i]; af[8 + i] = ahi[i]; }

      const v8bf blo = *(const v8bf*)(brow + e0);
      const v8bf bhi = *(const v8bf*)(brow + e0 + 16);
      v16bf bfr;
      #pragma unroll
      for (int i = 0; i < 8; ++i) { bfr[i] = blo[i]; bfr[8 + i] = bhi[i]; }

      acc = __builtin_amdgcn_wmma_f32_16x16x32_bf16(
          false, af, false, bfr, (short)0, acc, false, false);
    }
    __syncthreads();
  }

  // ---- split-K accumulate: C layout -> m = wave*16 + grp*8 + r, n = n0+lm ----
  const int n    = n0 + lm;
  const int mtop = (wave << 4) + (grp << 3);
  #pragma unroll
  for (int r = 0; r < 8; ++r) {
    unsafeAtomicAdd(&H1acc[(mtop + r) * D1_ + n], acc[r]);
  }
}

// ---------------------------------------------------------------------------
// Kernel 1 (fallback, if ws too small for the bf16 table): gather f32
// embeddings, mask-multiply + convert in registers. Same tiling/atomics.
// ---------------------------------------------------------------------------
__global__ __launch_bounds__(512) void gemm1_fused(
    const int*   __restrict__ data,  const int*   __restrict__ mask,
    const float* __restrict__ emb,   const float* __restrict__ W1,
    float*       __restrict__ H1acc) {
  __shared__ __bf16 Bs[16 * EE];

  const int tid  = threadIdx.x;
  const int lane = tid & 31;
  const int wave = tid >> 5;
  const int n0   = blockIdx.x << 4;
  const int t0   = blockIdx.y << 3;
  const int lm   = lane & 15;
  const int grp  = lane >> 4;
  const int eoff = grp << 3;
  const int mrow = (wave << 4) + lm;

  v8f acc = {};

  for (int t = t0; t < t0 + 8; ++t) {
    stage_w1(W1 + (size_t)(n0 + wave) * K1_ + (size_t)t * EE,
             Bs + wave * EE, lane);
    __syncthreads();

    const int   tok = data[mrow * TT_ + t];
    const int   mk  = mask[mrow * TT_ + t];
    const float fm  = (float)mk;
    const float* erow = emb + (size_t)(tok * mk) * EE;
    const __bf16* brow = Bs + lm * EE;

    #pragma unroll 4
    for (int es = 0; es < 20; ++es) {
      const int e0 = (es << 5) + eoff;
      const float4 a0 = *(const float4*)(erow + e0);
      const float4 a1 = *(const float4*)(erow + e0 + 4);
      const float4 a2 = *(const float4*)(erow + e0 + 16);
      const float4 a3 = *(const float4*)(erow + e0 + 20);
      v16bf af;
      af[0]  = (__bf16)(a0.x * fm); af[1]  = (__bf16)(a0.y * fm);
      af[2]  = (__bf16)(a0.z * fm); af[3]  = (__bf16)(a0.w * fm);
      af[4]  = (__bf16)(a1.x * fm); af[5]  = (__bf16)(a1.y * fm);
      af[6]  = (__bf16)(a1.z * fm); af[7]  = (__bf16)(a1.w * fm);
      af[8]  = (__bf16)(a2.x * fm); af[9]  = (__bf16)(a2.y * fm);
      af[10] = (__bf16)(a2.z * fm); af[11] = (__bf16)(a2.w * fm);
      af[12] = (__bf16)(a3.x * fm); af[13] = (__bf16)(a3.y * fm);
      af[14] = (__bf16)(a3.z * fm); af[15] = (__bf16)(a3.w * fm);

      const v8bf blo = *(const v8bf*)(brow + e0);
      const v8bf bhi = *(const v8bf*)(brow + e0 + 16);
      v16bf bfr;
      #pragma unroll
      for (int i = 0; i < 8; ++i) { bfr[i] = blo[i]; bfr[8 + i] = bhi[i]; }

      acc = __builtin_amdgcn_wmma_f32_16x16x32_bf16(
          false, af, false, bfr, (short)0, acc, false, false);
    }
    __syncthreads();
  }

  const int n    = n0 + lm;
  const int mtop = (wave << 4) + (grp << 3);
  #pragma unroll
  for (int r = 0; r < 8; ++r) {
    unsafeAtomicAdd(&H1acc[(mtop + r) * D1_ + n], acc[r]);
  }
}

// ---------------------------------------------------------------------------
// Kernel 2: ReLU + convert layer-1 result to bf16 for the next WMMA GEMM.
// ---------------------------------------------------------------------------
__global__ void relu_cvt(const float* __restrict__ H1acc, __bf16* __restrict__ H1bf) {
  int i = blockIdx.x * blockDim.x + threadIdx.x;
  float v = H1acc[i];
  H1bf[i] = (__bf16)(v > 0.f ? v : 0.f);
}

// ---------------------------------------------------------------------------
// Kernel 3: GEMM2 [256,512](bf16) x W2[512,512](f32->bf16) + b2, ReLU -> f32.
// ---------------------------------------------------------------------------
__global__ __launch_bounds__(256) void gemm2(
    const __bf16* __restrict__ A,  const float* __restrict__ W2,
    const float*  __restrict__ b2, float*       __restrict__ H2) {
  const int tid  = threadIdx.x;
  const int lane = tid & 31;
  const int wave = tid >> 5;
  const int tile = (blockIdx.x << 3) + wave;   // 0..511
  const int m0   = (tile >> 5) << 4;
  const int n0   = (tile & 31) << 4;
  const int lm   = lane & 15;
  const int grp  = lane >> 4;

  const __bf16* arow = A  + (m0 + lm) * D1_;
  const float*  brow = W2 + (size_t)(n0 + lm) * D1_;

  v8f acc = {};
  #pragma unroll
  for (int ks = 0; ks < 16; ++ks) {
    const int k0 = (ks << 5) + (grp << 3);
    const v8bf alo = *(const v8bf*)(arow + k0);
    const v8bf ahi = *(const v8bf*)(arow + k0 + 16);
    v16bf af;
    #pragma unroll
    for (int i = 0; i < 8; ++i) { af[i] = alo[i]; af[8 + i] = ahi[i]; }

    const float4 w0  = *(const float4*)(brow + k0);
    const float4 w1  = *(const float4*)(brow + k0 + 4);
    const float4 w2v = *(const float4*)(brow + k0 + 16);
    const float4 w3  = *(const float4*)(brow + k0 + 20);
    v16bf bfr;
    bfr[0]  = (__bf16)w0.x;  bfr[1]  = (__bf16)w0.y;
    bfr[2]  = (__bf16)w0.z;  bfr[3]  = (__bf16)w0.w;
    bfr[4]  = (__bf16)w1.x;  bfr[5]  = (__bf16)w1.y;
    bfr[6]  = (__bf16)w1.z;  bfr[7]  = (__bf16)w1.w;
    bfr[8]  = (__bf16)w2v.x; bfr[9]  = (__bf16)w2v.y;
    bfr[10] = (__bf16)w2v.z; bfr[11] = (__bf16)w2v.w;
    bfr[12] = (__bf16)w3.x;  bfr[13] = (__bf16)w3.y;
    bfr[14] = (__bf16)w3.z;  bfr[15] = (__bf16)w3.w;

    acc = __builtin_amdgcn_wmma_f32_16x16x32_bf16(
        false, af, false, bfr, (short)0, acc, false, false);
  }

  const int   n    = n0 + lm;
  const float bias = b2[n];
  #pragma unroll
  for (int r = 0; r < 8; ++r) {
    const int m = m0 + (grp << 3) + r;
    float v = acc[r] + bias;
    H2[m * D2_ + n] = v > 0.f ? v : 0.f;
  }
}

// ---------------------------------------------------------------------------
// Kernel 4: classifier head [256,512] x Wp[2,512] + bp -> out[256,2] (f32).
// ---------------------------------------------------------------------------
__global__ void head(const float* __restrict__ H2, const float* __restrict__ Wp,
                     const float* __restrict__ bp, float* __restrict__ out) {
  const int idx = blockIdx.x * blockDim.x + threadIdx.x;  // 0..511
  const int m = idx >> 1, c = idx & 1;
  const float4* h = (const float4*)(H2 + m * D2_);
  const float4* w = (const float4*)(Wp + c * D2_);
  float s = bp[c];
  #pragma unroll 8
  for (int k = 0; k < D2_ / 4; ++k) {
    const float4 a = h[k], b = w[k];
    s += a.x * b.x + a.y * b.y + a.z * b.z + a.w * b.w;
  }
  out[idx] = s;
}

// ---------------------------------------------------------------------------
extern "C" void kernel_launch(void* const* d_in, const int* in_sizes, int n_in,
                              void* d_out, int out_size, void* d_ws, size_t ws_size,
                              hipStream_t stream) {
  const int*   data = (const int*)d_in[0];
  const int*   mask = (const int*)d_in[1];
  const float* emb  = (const float*)d_in[2];
  const float* W1   = (const float*)d_in[3];
  const float* b1   = (const float*)d_in[4];
  const float* W2   = (const float*)d_in[5];
  const float* b2   = (const float*)d_in[6];
  const float* Wp   = (const float*)d_in[7];
  const float* bp   = (const float*)d_in[8];
  float* out = (float*)d_out;

  const size_t embbf_bytes = (size_t)(VV + 1) * EE * sizeof(__bf16); // ~61 MB
  const size_t small_bytes = (512u << 10) + (256u << 10) + (512u << 10);
  char* ws = (char*)d_ws;

  if (ws_size >= embbf_bytes + small_bytes) {
    // Fast path: bf16 embedding table + zero row in workspace.
    __bf16* embbf = (__bf16*)ws;
    float*  H1acc = (float*)(ws + embbf_bytes);
    __bf16* H1bf  = (__bf16*)(ws + embbf_bytes + (512u << 10));
    float*  H2    = (float*)(ws + embbf_bytes + (512u << 10) + (256u << 10));

    cvt_emb  <<<(VV * EE) / (256 * 4) + 1, 256, 0, stream>>>(emb, embbf);
    init_h1  <<<512, 256, 0, stream>>>(H1acc, b1);
    gemm1_fast<<<dim3(32, 64), 512, 0, stream>>>(data, mask, embbf, W1, H1acc);
    relu_cvt <<<512, 256, 0, stream>>>(H1acc, H1bf);
    gemm2    <<<64, 256, 0, stream>>>(H1bf, W2, b2, H2);
    head     <<<2, 256, 0, stream>>>(H2, Wp, bp, out);
  } else {
    // Fallback: in-register mask-multiply + f32->bf16 convert.
    float*  H1acc = (float*)ws;
    __bf16* H1bf  = (__bf16*)(ws + (512u << 10));
    float*  H2    = (float*)(ws + (512u << 10) + (256u << 10));

    init_h1   <<<512, 256, 0, stream>>>(H1acc, b1);
    gemm1_fused<<<dim3(32, 64), 512, 0, stream>>>(data, mask, emb, W1, H1acc);
    relu_cvt  <<<512, 256, 0, stream>>>(H1acc, H1bf);
    gemm2     <<<64, 256, 0, stream>>>(H1bf, W2, b2, H2);
    head      <<<2, 256, 0, stream>>>(H2, Wp, bp, out);
  }
}
